// PointNetSetUpConv_9354438770947
// MI455X (gfx1250) — matrix operand order repaired and checked by reading, not compiled, not run on hardware
//
#include <hip/hip_runtime.h>
#include <hip/hip_bf16.h>

typedef _Float16 v8h  __attribute__((ext_vector_type(8)));
typedef _Float16 v16h __attribute__((ext_vector_type(16)));
typedef float    v8f  __attribute__((ext_vector_type(8)));
typedef int      v4i  __attribute__((ext_vector_type(4)));

#define BB   4
#define N1_  8192
#define N2_  2048
#define KNN  8
#define C1_  256
#define C2_  512
// MLP1: 515 -> 128 -> 128 -> 256 ; MLP2: 512 -> 256
#define CIN0 515
#define CPAD 544            // 515 padded to 544 (16B-aligned f16 rows)
#define H0   128
#define H1   128
#define H2   256
#define CIN2 512
#define COUT 256

#if defined(__has_builtin)
#if __has_builtin(__builtin_amdgcn_global_load_async_to_lds_b128)
#define HAVE_ASYNC 1
#endif
#endif
#ifndef HAVE_ASYNC
#define HAVE_ASYNC 0
#endif

// 16 f16 (2 x b128) global -> LDS
static __device__ __forceinline__ void cp16h(const _Float16* g, _Float16* l) {
#if HAVE_ASYNC
  __builtin_amdgcn_global_load_async_to_lds_b128(
      (__attribute__((address_space(1))) v4i*)g,
      (__attribute__((address_space(3))) v4i*)l, 0, 0);
  __builtin_amdgcn_global_load_async_to_lds_b128(
      (__attribute__((address_space(1))) v4i*)(g + 8),
      (__attribute__((address_space(3))) v4i*)(l + 8), 0, 0);
#else
  *(v8h*)l       = *(const v8h*)g;
  *(v8h*)(l + 8) = *(const v8h*)(g + 8);
#endif
}
// 8 f16 (1 x b128) global -> LDS
static __device__ __forceinline__ void cp8h(const _Float16* g, _Float16* l) {
#if HAVE_ASYNC
  __builtin_amdgcn_global_load_async_to_lds_b128(
      (__attribute__((address_space(1))) v4i*)g,
      (__attribute__((address_space(3))) v4i*)l, 0, 0);
#else
  *(v8h*)l = *(const v8h*)g;
#endif
}
static __device__ __forceinline__ void wait_async() {
#if HAVE_ASYNC
  asm volatile("s_wait_asynccnt 0" ::: "memory");
#endif
}

// ---- WMMA fragment loaders (LDS, f16) ----
static __device__ __forceinline__ v16h ldA(const _Float16* p) {
  v8h lo = *(const v8h*)(p);
  v8h hi = *(const v8h*)(p + 8);
  v16h r;
#pragma unroll
  for (int i = 0; i < 8; ++i) { r[i] = lo[i]; r[i + 8] = hi[i]; }
  return r;
}
static __device__ __forceinline__ v16h ldB(const _Float16* p) {
  v8h lo = *(const v8h*)(p);
  v8h hi = *(const v8h*)(p + 8);
  v16h r;
#pragma unroll
  for (int i = 0; i < 8; ++i) { r[i] = lo[i]; r[i + 8] = hi[i]; }
  return r;
}
static __device__ __forceinline__ float relu_f(float v) { return v > 0.f ? v : 0.f; }

// =====================================================================
// Kernel P: pre-convert all weights to f16 (w0 zero-padded to 544 cols)
// =====================================================================
__global__ __launch_bounds__(256) void prep_weights_kernel(
    const float* __restrict__ w0, const float* __restrict__ w1,
    const float* __restrict__ w2, const float* __restrict__ w3,
    _Float16* __restrict__ w0h, _Float16* __restrict__ w1h,
    _Float16* __restrict__ w2h, _Float16* __restrict__ w3h) {
  const int t0 = blockIdx.x * 256 + threadIdx.x;
  const int stride = gridDim.x * 256;
  for (int i = t0; i < H0 * CPAD; i += stride) {
    int o = i / CPAD, c = i % CPAD;
    w0h[i] = (c < CIN0) ? (_Float16)w0[o * CIN0 + c] : (_Float16)0.f;
  }
  for (int i = t0; i < H1 * H0; i += stride)   w1h[i] = (_Float16)w1[i];
  for (int i = t0; i < H2 * H1; i += stride)   w2h[i] = (_Float16)w2[i];
  for (int i = t0; i < COUT * CIN2; i += stride) w3h[i] = (_Float16)w3[i];
}

// =====================================================================
// Kernel T2: feature2 (B,C2,N2) ++ pos2 ++ pad -> feat2T (B,N2,544) f16
// =====================================================================
__global__ __launch_bounds__(256) void transpose_f2_kernel(
    const float* __restrict__ feature2, const float* __restrict__ pos2,
    _Float16* __restrict__ feat2T) {
  __shared__ float tile[32][33];
  const int tid = threadIdx.x;
  const int tx = tid & 31, ty = tid >> 5;
  const int nBase = blockIdx.x * 32;
  const int ct = blockIdx.y;                // 0..16
  const int b  = blockIdx.z;
#pragma unroll
  for (int j = 0; j < 4; ++j) {
    int cl = ty + j * 8;
    int ch = ct * 32 + cl;
    int n  = nBase + tx;
    float v;
    if (ch < C2_)       v = feature2[((b * C2_) + ch) * N2_ + n];
    else if (ch < CIN0) v = pos2[((b * 3) + (ch - C2_)) * N2_ + n];
    else                v = 0.f;
    tile[cl][tx] = v;
  }
  __syncthreads();
#pragma unroll
  for (int j = 0; j < 4; ++j) {
    int nl = ty + j * 8;
    feat2T[((size_t)(b * N2_) + (nBase + nl)) * CPAD + ct * 32 + tx] =
        (_Float16)tile[tx][nl];
  }
}

// =====================================================================
// Kernel T1: feature1 (B,C1,N1) -> cat[b][n][256..511] f16
// =====================================================================
__global__ __launch_bounds__(256) void transpose_f1_kernel(
    const float* __restrict__ feature1, _Float16* __restrict__ cat) {
  __shared__ float tile[32][33];
  const int tid = threadIdx.x;
  const int tx = tid & 31, ty = tid >> 5;
  const int nBase = blockIdx.x * 32;
  const int ct = blockIdx.y;                // 0..7
  const int b  = blockIdx.z;
#pragma unroll
  for (int j = 0; j < 4; ++j) {
    int cl = ty + j * 8;
    tile[cl][tx] = feature1[((b * C1_) + (ct * 32 + cl)) * N1_ + nBase + tx];
  }
  __syncthreads();
#pragma unroll
  for (int j = 0; j < 4; ++j) {
    int nl = ty + j * 8;
    cat[((size_t)(b * N1_) + (nBase + nl)) * CIN2 + H2 + ct * 32 + tx] =
        (_Float16)tile[tx][nl];
  }
}

// =====================================================================
// Kernel 1: KNN (K=8), 1 thread/query, support points in LDS.
// =====================================================================
__global__ __launch_bounds__(256) void knn_kernel(
    const float* __restrict__ pos1, const float* __restrict__ pos2,
    int* __restrict__ idx_out) {
  __shared__ float sp[3 * N2_];
  const int tid = threadIdx.x;
  const int b   = blockIdx.y;
#pragma unroll
  for (int i = 0; i < (3 * N2_) / 256; ++i) {
    int lin = i * 256 + tid;
    sp[lin] = pos2[b * 3 * N2_ + lin];
  }
  __syncthreads();

  const int n  = blockIdx.x * 256 + tid;
  const float qx = pos1[(b * 3 + 0) * N1_ + n];
  const float qy = pos1[(b * 3 + 1) * N1_ + n];
  const float qz = pos1[(b * 3 + 2) * N1_ + n];

  float bd[KNN]; int bi[KNN];
#pragma unroll
  for (int k = 0; k < KNN; ++k) { bd[k] = 3.4e38f; bi[k] = 0; }

  for (int j = 0; j < N2_; ++j) {
    float dx = sp[j] - qx, dy = sp[N2_ + j] - qy, dz = sp[2 * N2_ + j] - qz;
    float d  = dx * dx + dy * dy + dz * dz;
    if (d < bd[KNN - 1]) {
      bd[KNN - 1] = d; bi[KNN - 1] = j;
#pragma unroll
      for (int t = KNN - 1; t > 0; --t) {
        if (bd[t] < bd[t - 1]) {
          float td = bd[t]; bd[t] = bd[t - 1]; bd[t - 1] = td;
          int   ti = bi[t]; bi[t] = bi[t - 1]; bi[t - 1] = ti;
        }
      }
    }
  }
#pragma unroll
  for (int k = 0; k < KNN; ++k)
    idx_out[((b * N1_) + n) * KNN + k] = bi[k];
}

// =====================================================================
// Kernel 2: fused gather + MLP1 (515->128->128->256) + max over 8 nbrs.
// 128 rows (16 queries x 8 nbrs), 8 waves x 16-row tiles.
// Double-buffered async staging: issue chunk k+1 after the barrier,
// compute chunk k while the DMA runs.
// =====================================================================
__global__ __launch_bounds__(256) void mlp1_kernel(
    const float* __restrict__ pos1, const _Float16* __restrict__ feat2T,
    const int* __restrict__ nbr_idx,
    const _Float16* __restrict__ w0h, const float* __restrict__ s0, const float* __restrict__ b0,
    const _Float16* __restrict__ w1h, const float* __restrict__ s1, const float* __restrict__ b1,
    const _Float16* __restrict__ w2h, const float* __restrict__ s2, const float* __restrict__ b2,
    _Float16* __restrict__ cat) {
  __shared__ _Float16 sFeat[2][128 * 32];
  __shared__ _Float16 sW[2][128 * 32];
  __shared__ _Float16 sAct1[128 * H0];
  __shared__ _Float16 sAct2[128 * H1];
  __shared__ int      sIdx[128];

  const int tid   = threadIdx.x;
  const int lane  = tid & 31;
  const int wv    = tid >> 5;
  const int b     = blockIdx.y;
  const int qBase = blockIdx.x * 16;
  const int hl    = lane >> 4;
  const int ln    = lane & 15;
  const int r128  = tid & 127;
  const int cHalf = (tid >> 7) << 4;  // 0 or 16

  if (tid < 128) {
    int q = qBase + (tid >> 3), j = tid & 7;
    sIdx[tid] = nbr_idx[((b * N1_) + q) * KNN + j];
  }
  __syncthreads();
  const int pi = sIdx[r128];
  const _Float16* rowp = feat2T + ((size_t)(b * N2_) + pi) * CPAD;
  float p1v[3];
  {
    int q = qBase + (r128 >> 3);
#pragma unroll
    for (int d = 0; d < 3; ++d) p1v[d] = pos1[(b * 3 + d) * N1_ + q];
  }

  // ---------------- Layer 1: 515 -> 128 (17 K-chunks, double-buffered) --
  v8f acc[8];
#pragma unroll
  for (int ct = 0; ct < 8; ++ct) acc[ct] = v8f{0.f,0.f,0.f,0.f,0.f,0.f,0.f,0.f};

  // prologue: chunk 0 into buffer 0
  cp16h(rowp + cHalf, &sFeat[0][r128 * 32 + cHalf]);
  cp16h(w0h + r128 * CPAD + cHalf, &sW[0][r128 * 32 + cHalf]);

  for (int kc = 0; kc < 17; ++kc) {
    const int cur = kc & 1, nxt = cur ^ 1;
    wait_async();
    __syncthreads();                 // buf[cur] ready; buf[nxt] readers done
    if (kc + 1 < 17) {
      const int cb = (kc + 1) * 32;
      if (kc + 1 < 16) {
        cp16h(rowp + cb + cHalf, &sFeat[nxt][r128 * 32 + cHalf]);
      } else {                       // chunk 16: pos_diff fixup (sync path)
        v8h x0 = *(const v8h*)(rowp + cb + cHalf);
        v8h x1 = *(const v8h*)(rowp + cb + cHalf + 8);
        if (cHalf == 0) {
          x0[0] = (_Float16)((float)x0[0] - p1v[0]);
          x0[1] = (_Float16)((float)x0[1] - p1v[1]);
          x0[2] = (_Float16)((float)x0[2] - p1v[2]);
        }
        _Float16* dst = &sFeat[nxt][r128 * 32 + cHalf];
        *(v8h*)dst = x0; *(v8h*)(dst + 8) = x1;
      }
      cp16h(w0h + r128 * CPAD + cb + cHalf, &sW[nxt][r128 * 32 + cHalf]);
    }
    v16h a = ldA(&sFeat[cur][(wv * 16 + ln) * 32 + hl * 16]);
#pragma unroll
    for (int ct = 0; ct < 8; ++ct) {
      v16h bm = ldB(&sW[cur][(ct * 16 + ln) * 32 + hl * 16]);
      acc[ct] = __builtin_amdgcn_wmma_f32_16x16x32_f16(
          false, a, false, bm, (short)0, acc[ct], false, false);
    }
  }
#pragma unroll
  for (int ct = 0; ct < 8; ++ct) {
    int ch = ct * 16 + ln;
    float sc = s0[ch], bi_ = b0[ch];
#pragma unroll
    for (int i = 0; i < 8; ++i) {
      float v = relu_f(acc[ct][i] * sc + bi_);
      sAct1[(wv * 16 + i + (hl << 3)) * H0 + ch] = (_Float16)v;
    }
  }
  __syncthreads();

  // ---------------- Layer 2: 128 -> 128 (8 ct-tiles, double-buffered) ---
  const int wo = tid >> 4;          // 0..15 out row for weight staging
  const int wc = (tid & 15) * 8;    // 0..120 col
  cp8h(w1h + wo * H0 + wc, &sW[0][wo * H0 + wc]);
  for (int ct = 0; ct < 8; ++ct) {
    const int cur = ct & 1, nxt = cur ^ 1;
    wait_async();
    __syncthreads();
    if (ct + 1 < 8)
      cp8h(w1h + ((ct + 1) * 16 + wo) * H0 + wc, &sW[nxt][wo * H0 + wc]);
    v8f a2 = v8f{0.f,0.f,0.f,0.f,0.f,0.f,0.f,0.f};
#pragma unroll
    for (int kc = 0; kc < 4; ++kc) {
      v16h a  = ldA(&sAct1[(wv * 16 + ln) * H0 + kc * 32 + hl * 16]);
      v16h bm = ldB(&sW[cur][ln * H0 + kc * 32 + hl * 16]);
      a2 = __builtin_amdgcn_wmma_f32_16x16x32_f16(
          false, a, false, bm, (short)0, a2, false, false);
    }
    int ch = ct * 16 + ln;
    float sc = s1[ch], bi_ = b1[ch];
#pragma unroll
    for (int i = 0; i < 8; ++i) {
      float v = relu_f(a2[i] * sc + bi_);
      sAct2[(wv * 16 + i + (hl << 3)) * H1 + ch] = (_Float16)v;
    }
  }
  __syncthreads();

  // ------------- Layer 3: 128 -> 256, fused max over neighbors ----------
  cp8h(w2h + wo * H1 + wc, &sW[0][wo * H1 + wc]);
  for (int ct = 0; ct < 16; ++ct) {
    const int cur = ct & 1, nxt = cur ^ 1;
    wait_async();
    __syncthreads();
    if (ct + 1 < 16)
      cp8h(w2h + ((ct + 1) * 16 + wo) * H1 + wc, &sW[nxt][wo * H1 + wc]);
    v8f a3 = v8f{0.f,0.f,0.f,0.f,0.f,0.f,0.f,0.f};
#pragma unroll
    for (int kc = 0; kc < 4; ++kc) {
      v16h a  = ldA(&sAct2[(wv * 16 + ln) * H1 + kc * 32 + hl * 16]);
      v16h bm = ldB(&sW[cur][ln * H1 + kc * 32 + hl * 16]);
      a3 = __builtin_amdgcn_wmma_f32_16x16x32_f16(
          false, a, false, bm, (short)0, a3, false, false);
    }
    // 8 accumulator elements of one lane-half = the 8 neighbors of a query
    int ch = ct * 16 + ln;
    float sc = s2[ch], bi_ = b2[ch];
    float mv = 0.f;                 // relu output >= 0
#pragma unroll
    for (int i = 0; i < 8; ++i)
      mv = fmaxf(mv, relu_f(a3[i] * sc + bi_));
    cat[((size_t)(b * N1_) + (qBase + 2 * wv + hl)) * CIN2 + ch] = (_Float16)mv;
  }
}

// =====================================================================
// Kernel 3: cat (B,N1,512) f16 -> 256 outs, write (B,256,N1) f32.
// Fully async double-buffered staging (input rows + weights).
// =====================================================================
__global__ __launch_bounds__(256) void mlp2_kernel(
    const _Float16* __restrict__ cat, const _Float16* __restrict__ w3h,
    const float* __restrict__ s, const float* __restrict__ bia,
    float* __restrict__ out) {
  __shared__ _Float16 sIn[2][128 * 32];
  __shared__ _Float16 sW[2][128 * 32];

  const int tid   = threadIdx.x;
  const int lane  = tid & 31;
  const int wv    = tid >> 5;
  const int b     = blockIdx.y;
  const int nBase = blockIdx.x * 128;
  const int hl    = lane >> 4;
  const int ln    = lane & 15;
  const int r128  = tid & 127;
  const int cHalf = (tid >> 7) << 4;
  const _Float16* crow = cat + ((size_t)(b * N1_) + nBase + r128) * CIN2;

  for (int half = 0; half < 2; ++half) {
    const _Float16* wbase = w3h + (size_t)(half * 128 + r128) * CIN2;
    v8f acc[8];
#pragma unroll
    for (int ct = 0; ct < 8; ++ct) acc[ct] = v8f{0.f,0.f,0.f,0.f,0.f,0.f,0.f,0.f};

    cp16h(crow + cHalf,  &sIn[0][r128 * 32 + cHalf]);
    cp16h(wbase + cHalf, &sW[0][r128 * 32 + cHalf]);

    for (int kc = 0; kc < 16; ++kc) {
      const int cur = kc & 1, nxt = cur ^ 1;
      wait_async();
      __syncthreads();
      if (kc + 1 < 16) {
        const int cb = (kc + 1) * 32;
        cp16h(crow + cb + cHalf,  &sIn[nxt][r128 * 32 + cHalf]);
        cp16h(wbase + cb + cHalf, &sW[nxt][r128 * 32 + cHalf]);
      }
      v16h a = ldA(&sIn[cur][(wv * 16 + ln) * 32 + hl * 16]);
#pragma unroll
      for (int ct = 0; ct < 8; ++ct) {
        v16h bm = ldB(&sW[cur][(ct * 16 + ln) * 32 + hl * 16]);
        acc[ct] = __builtin_amdgcn_wmma_f32_16x16x32_f16(
            false, a, false, bm, (short)0, acc[ct], false, false);
      }
    }
#pragma unroll
    for (int ct = 0; ct < 8; ++ct) {
      int o = half * 128 + ct * 16 + ln;
      float sc = s[o], bi_ = bia[o];
#pragma unroll
      for (int i = 0; i < 8; ++i) {
        float v = relu_f(acc[ct][i] * sc + bi_);
        out[((size_t)(b * COUT) + o) * N1_ + nBase + wv * 16 + i + (hl << 3)] = v;
      }
    }
    __syncthreads();   // before next half's prologue reuses buffer 0
  }
}

// =====================================================================
extern "C" void kernel_launch(void* const* d_in, const int* in_sizes, int n_in,
                              void* d_out, int out_size, void* d_ws, size_t ws_size,
                              hipStream_t stream) {
  const float* pos1     = (const float*)d_in[0];
  const float* pos2     = (const float*)d_in[1];
  const float* feature1 = (const float*)d_in[2];
  const float* feature2 = (const float*)d_in[3];
  const float* w1_0 = (const float*)d_in[4];
  const float* s1_0 = (const float*)d_in[5];
  const float* b1_0 = (const float*)d_in[6];
  const float* w1_1 = (const float*)d_in[7];
  const float* s1_1 = (const float*)d_in[8];
  const float* b1_1 = (const float*)d_in[9];
  const float* w1_2 = (const float*)d_in[10];
  const float* s1_2 = (const float*)d_in[11];
  const float* b1_2 = (const float*)d_in[12];
  const float* w2_0 = (const float*)d_in[13];
  const float* s2_0 = (const float*)d_in[14];
  const float* b2_0 = (const float*)d_in[15];
  float* out = (float*)d_out;

  // ws: [idx][cat B*N1*512 f16][feat2T B*N2*544 f16][w0h][w1h][w2h][w3h]
  char* wsp = (char*)d_ws;
  int* idx_ws = (int*)wsp;
  size_t off = (size_t)BB * N1_ * KNN * sizeof(int);
  _Float16* cat = (_Float16*)(wsp + off);
  off += (size_t)BB * N1_ * CIN2 * sizeof(_Float16);
  _Float16* feat2T = (_Float16*)(wsp + off);
  off += (size_t)BB * N2_ * CPAD * sizeof(_Float16);
  _Float16* w0h = (_Float16*)(wsp + off); off += (size_t)H0 * CPAD * sizeof(_Float16);
  _Float16* w1h = (_Float16*)(wsp + off); off += (size_t)H1 * H0 * sizeof(_Float16);
  _Float16* w2h = (_Float16*)(wsp + off); off += (size_t)H2 * H1 * sizeof(_Float16);
  _Float16* w3h = (_Float16*)(wsp + off); off += (size_t)COUT * CIN2 * sizeof(_Float16);

  prep_weights_kernel<<<dim3(256), 256, 0, stream>>>(
      w1_0, w1_1, w1_2, w2_0, w0h, w1h, w2h, w3h);
  transpose_f2_kernel<<<dim3(N2_ / 32, 17, BB), 256, 0, stream>>>(
      feature2, pos2, feat2T);
  transpose_f1_kernel<<<dim3(N1_ / 32, C1_ / 32, BB), 256, 0, stream>>>(
      feature1, cat);
  knn_kernel<<<dim3(N1_ / 256, BB), 256, 0, stream>>>(pos1, pos2, idx_ws);
  mlp1_kernel<<<dim3(N1_ / 16, BB), 256, 0, stream>>>(
      pos1, feat2T, idx_ws,
      w0h, s1_0, b1_0, w1h, s1_1, b1_1, w2h, s1_2, b1_2, cat);
  mlp2_kernel<<<dim3(N1_ / 128, BB), 256, 0, stream>>>(
      cat, w3h, s2_0, b2_0, out);
}
